// SchNetBlock_67439576482320
// MI455X (gfx1250) — matrix operand order repaired and checked by reading, not compiled, not run on hardware
//
#include <hip/hip_runtime.h>

#define NF 128      // HIDDEN == NUM_FILTERS == 128
#define PI_F 3.14159265358979f

typedef __attribute__((ext_vector_type(16))) __bf16 v16bf;
typedef __attribute__((ext_vector_type(8)))  float  v8f;

#define WMMA_BF16(A, B, C) \
    __builtin_amdgcn_wmma_f32_16x16x32_bf16(false, (A), false, (B), (short)0, (C), false, false)

__device__ __forceinline__ float ssp(float v) {        // shifted softplus
    float s = (v > 20.0f) ? v : __logf(1.0f + __expf(v));
    return s - 0.69314718055994531f;
}

// ---------- fragment-order index helpers (bf16, contiguous k = 16*(lane>>4) + idx) ----------
// B frag (32x16 K-major tile): element idx of lane -> k = ((lane>>4)<<4) + idx.
// Stored pre-swizzled: frag[(kc*8+nt)*32 + lane][16] contiguous per lane.
__device__ __forceinline__ v16bf load_B_frag(const __bf16* F, int kc, int nt, int lane) {
    return *reinterpret_cast<const v16bf*>(F + ((((kc << 3) + nt) << 5) + lane) * 16);
}
// A frag inverse map: element (m, khat) khat=0..31 lives at
//   lane' = m + 16*((khat>>4)&1) ,  idx = khat&15
__device__ __forceinline__ int afrag_lane(int m, int khat) { return m + (((khat >> 4) & 1) << 4); }
__device__ __forceinline__ int afrag_idx(int khat)         { return khat & 15; }

// -------------------- weight pre-swizzle: f32 [K,128] -> bf16 fragments --------------------
__global__ void prep_wfrag_kernel(const float* __restrict__ W, __bf16* __restrict__ F,
                                  int Kreal, int Kc) {
    int total = Kc * 8 * 32 * 16;
    for (int i = blockIdx.x * blockDim.x + threadIdx.x; i < total;
         i += gridDim.x * blockDim.x) {
        int idx  = i & 15;
        int lane = (i >> 4) & 31;
        int tile = i >> 9;              // kc*8 + nt
        int nt = tile & 7, kc = tile >> 3;
        int k = kc * 32 + ((lane >> 4) << 4) + idx;
        int n = nt * 16 + (lane & 15);
        F[i] = (k < Kreal) ? (__bf16)W[k * NF + n] : (__bf16)0.0f;
    }
}

__global__ void zero_kernel(float* __restrict__ p, long long n) {
    long long i = (long long)blockIdx.x * blockDim.x + threadIdx.x;
    long long s = (long long)gridDim.x * blockDim.x;
    for (; i < n; i += s) p[i] = 0.0f;
}

// ---------- x[Mpad,128] = A[M,128] @ Wfrag ; out rows padded, stores unguarded ----------
__global__ void __launch_bounds__(128) gemm128_kernel(
    const float* __restrict__ A, const __bf16* __restrict__ Wf,
    float* __restrict__ out, int M)
{
    __shared__ alignas(32) __bf16 at[4][4][32][16];   // per wave, A-frag order, 16 KB
    const int tid = threadIdx.x, lane = tid & 31, w = tid >> 5;
    const int m0 = (blockIdx.x * 4 + w) * 16;

    for (int i = lane; i < 16 * NF; i += 32) {        // element (m,k)
        int m = i >> 7, k = i & 127;
        int mr = min(m0 + m, M - 1);                  // clamp load row
        at[w][k >> 5][afrag_lane(m, k & 31)][afrag_idx(k & 31)] = (__bf16)A[(size_t)mr * NF + k];
    }
    __syncthreads();

    const int n = lane & 15, hb = (lane >> 4) << 3;
#pragma unroll
    for (int nt = 0; nt < 8; ++nt) {
        v8f acc = {};
#pragma unroll
        for (int kc = 0; kc < 4; ++kc) {
            v16bf a = *reinterpret_cast<const v16bf*>(&at[w][kc][lane][0]);
            acc = WMMA_BF16(a, load_B_frag(Wf, kc, nt, lane), acc);
        }
        float* o = out + (size_t)(m0 + hb) * NF + nt * 16 + n;   // padded, always in-range
#pragma unroll
        for (int r = 0; r < 8; ++r) o[(size_t)r * NF] = acc[r];
    }
}

// -------------------- fused per-edge CFConv filter + scatter --------------------
// Branch-free tail: edge index clamped, cutoff C forced to 0 for invalid edges,
// so the unconditional atomicAdd contributes exactly 0.
__global__ void __launch_bounds__(256) edge_kernel(
    const float* __restrict__ pos, const int* __restrict__ eidx,
    const __bf16* __restrict__ w1f, const float* __restrict__ b1,
    const __bf16* __restrict__ w2f, const float* __restrict__ b2,
    const float* __restrict__ xfeat, float* __restrict__ agg,
    int nE, int NG)
{
    __shared__ alignas(32) __bf16 gaFrag[8][2][32][16];   // 16 KB
    __shared__ alignas(32) __bf16 hidFrag[8][4][32][16];  // 32 KB
    __shared__ float Cw[8][16];
    __shared__ int   rsrc[8][16], rdst[8][16];

    const int tid = threadIdx.x, lane = tid & 31, w = tid >> 5;
    const int e0 = (blockIdx.x * 8 + w) * 16;
    const int m  = lane & 15;
    const bool valid = (e0 + m) < nE;
    const int e  = valid ? (e0 + m) : (nE - 1);           // clamped, always loadable

    const int rdest = eidx[e];
    const int csrc  = eidx[nE + e];
    float dx = pos[3 * rdest]     - pos[3 * csrc];
    float dy = pos[3 * rdest + 1] - pos[3 * csrc + 1];
    float dz = pos[3 * rdest + 2] - pos[3 * csrc + 2];
    const float d = sqrtf(dx * dx + dy * dy + dz * dz);

    if (lane < 16) {
        Cw[w][m]   = valid ? 0.5f * (__cosf(d * PI_F / 10.0f) + 1.0f) : 0.0f;
        rsrc[w][m] = csrc;
        rdst[w][m] = rdest;
    }
    {   // each lane-half fills one 32-wide K chunk of its edge's gaussians
        const float step  = 10.0f / (float)(NG - 1);
        const float coeff = -0.5f / (step * step);
        const int kc = lane >> 4;                 // 0 or 1
#pragma unroll
        for (int gg = 0; gg < 32; ++gg) {
            int g = kc * 32 + gg;
            float t = d - (float)g * step;
            float v = (g < NG) ? __expf(coeff * t * t) : 0.0f;
            gaFrag[w][kc][afrag_lane(m, gg)][afrag_idx(gg)] = (__bf16)v;
        }
    }
    __syncthreads();

    const int n = lane & 15, hb = (lane >> 4) << 3;

    // layer 1: [16 x 64] @ [64 x 128]
    v16bf a0 = *reinterpret_cast<const v16bf*>(&gaFrag[w][0][lane][0]);
    v16bf a1 = *reinterpret_cast<const v16bf*>(&gaFrag[w][1][lane][0]);
#pragma unroll
    for (int nt = 0; nt < 8; ++nt) {
        v8f acc = {};
        acc = WMMA_BF16(a0, load_B_frag(w1f, 0, nt, lane), acc);
        acc = WMMA_BF16(a1, load_B_frag(w1f, 1, nt, lane), acc);
        float bias = b1[nt * 16 + n];
        int filter = nt * 16 + n;
        int khat = filter & 31;
        int fl = afrag_lane(0, khat) + hb, fi = afrag_idx(khat);
#pragma unroll
        for (int r = 0; r < 8; ++r)
            hidFrag[w][filter >> 5][fl + r][fi] = (__bf16)ssp(acc[r] + bias);
    }
    __syncthreads();

    // layer 2: [16 x 128] @ [128 x 128]  -> cutoff * gather * scatter-add
#pragma unroll
    for (int nt = 0; nt < 8; ++nt) {
        v8f acc = {};
#pragma unroll
        for (int kc = 0; kc < 4; ++kc) {
            v16bf a = *reinterpret_cast<const v16bf*>(&hidFrag[w][kc][lane][0]);
            acc = WMMA_BF16(a, load_B_frag(w2f, kc, nt, lane), acc);
        }
        float bias = b2[nt * 16 + n];
#pragma unroll
        for (int r = 0; r < 8; ++r) {
            int mm = r + hb;
            float Wv  = (acc[r] + bias) * Cw[w][mm];            // 0 for invalid edges
            float msg = Wv * xfeat[(size_t)rsrc[w][mm] * NF + nt * 16 + n];
            atomicAdd(&agg[(size_t)rdst[w][mm] * NF + nt * 16 + n], msg);
        }
    }
}

// -------------------- out = ssp(agg@lin2 + b2) @ lin + b --------------------
__global__ void __launch_bounds__(128) out_kernel(
    const float* __restrict__ agg,
    const __bf16* __restrict__ waf, const float* __restrict__ ba,
    const __bf16* __restrict__ wbf, const float* __restrict__ bb,
    float* __restrict__ out, int M)
{
    __shared__ alignas(32) __bf16 at[4][4][32][16];    // 16 KB
    __shared__ alignas(32) __bf16 hid[4][4][32][16];   // 16 KB
    const int tid = threadIdx.x, lane = tid & 31, w = tid >> 5;
    const int m0 = (blockIdx.x * 4 + w) * 16;

    for (int i = lane; i < 16 * NF; i += 32) {
        int m = i >> 7, k = i & 127;
        int mr = min(m0 + m, M - 1);                  // agg is padded; clamp is safe+cheap
        at[w][k >> 5][afrag_lane(m, k & 31)][afrag_idx(k & 31)] = (__bf16)agg[(size_t)mr * NF + k];
    }
    __syncthreads();

    const int n = lane & 15, hb = (lane >> 4) << 3;
#pragma unroll
    for (int nt = 0; nt < 8; ++nt) {
        v8f acc = {};
#pragma unroll
        for (int kc = 0; kc < 4; ++kc) {
            v16bf a = *reinterpret_cast<const v16bf*>(&at[w][kc][lane][0]);
            acc = WMMA_BF16(a, load_B_frag(waf, kc, nt, lane), acc);
        }
        float bias = ba[nt * 16 + n];
        int filter = nt * 16 + n;
        int khat = filter & 31;
        int fl = afrag_lane(0, khat) + hb, fi = afrag_idx(khat);
#pragma unroll
        for (int r = 0; r < 8; ++r)
            hid[w][filter >> 5][fl + r][fi] = (__bf16)ssp(acc[r] + bias);
    }
    __syncthreads();

    const bool full = (m0 + 16 <= M);                 // wave-uniform; tail path is cold
#pragma unroll
    for (int nt = 0; nt < 8; ++nt) {
        v8f acc = {};
#pragma unroll
        for (int kc = 0; kc < 4; ++kc) {
            v16bf a = *reinterpret_cast<const v16bf*>(&hid[w][kc][lane][0]);
            acc = WMMA_BF16(a, load_B_frag(wbf, kc, nt, lane), acc);
        }
        float bias = bb[nt * 16 + n];
        float* o = out + (size_t)(m0 + hb) * NF + nt * 16 + n;
        if (full) {
#pragma unroll
            for (int r = 0; r < 8; ++r) o[(size_t)r * NF] = acc[r] + bias;
        } else {
#pragma unroll
            for (int r = 0; r < 8; ++r)
                if (m0 + hb + r < M) o[(size_t)r * NF] = acc[r] + bias;
        }
    }
}

extern "C" void kernel_launch(void* const* d_in, const int* in_sizes, int n_in,
                              void* d_out, int out_size, void* d_ws, size_t ws_size,
                              hipStream_t stream) {
    const float* h      = (const float*)d_in[0];
    const float* pos    = (const float*)d_in[1];
    const int*   eidx   = (const int*)  d_in[2];
    const float* mlp_w1 = (const float*)d_in[3];
    const float* mlp_b1 = (const float*)d_in[4];
    const float* mlp_w2 = (const float*)d_in[5];
    const float* mlp_b2 = (const float*)d_in[6];
    const float* lin1_w = (const float*)d_in[7];
    const float* lin2_w = (const float*)d_in[8];
    const float* lin2_b = (const float*)d_in[9];
    const float* lin_w  = (const float*)d_in[10];
    const float* lin_b  = (const float*)d_in[11];
    float* out = (float*)d_out;

    const int N  = in_sizes[0] / NF;
    const int E  = in_sizes[2] / 2;
    const int NG = in_sizes[3] / NF;
    const int Npad = (N + 63) & ~63;           // pad node rows to full 4-wave blocks

    // workspace layout (256B aligned slices); x/agg padded so stores are unguarded
    auto align256 = [](size_t v) { return (v + 255) & ~(size_t)255; };
    char* ws = (char*)d_ws;
    float* x    = (float*)ws;                 ws += align256((size_t)Npad * NF * 4);
    float* agg  = (float*)ws;                 ws += align256((size_t)Npad * NF * 4);
    __bf16* w1f  = (__bf16*)ws;               ws += align256((size_t)2 * 4096 * 2);
    __bf16* w2f  = (__bf16*)ws;               ws += align256((size_t)4 * 4096 * 2);
    __bf16* l1f  = (__bf16*)ws;               ws += align256((size_t)4 * 4096 * 2);
    __bf16* l2f  = (__bf16*)ws;               ws += align256((size_t)4 * 4096 * 2);
    __bf16* lf   = (__bf16*)ws;               ws += align256((size_t)4 * 4096 * 2);

    // pre-swizzle weights into bf16 WMMA B-fragment order
    prep_wfrag_kernel<<<8, 256, 0, stream>>>(mlp_w1, w1f, NG, 2);
    prep_wfrag_kernel<<<16, 256, 0, stream>>>(mlp_w2, w2f, NF, 4);
    prep_wfrag_kernel<<<16, 256, 0, stream>>>(lin1_w, l1f, NF, 4);
    prep_wfrag_kernel<<<16, 256, 0, stream>>>(lin2_w, l2f, NF, 4);
    prep_wfrag_kernel<<<16, 256, 0, stream>>>(lin_w,  lf,  NF, 4);

    zero_kernel<<<512, 256, 0, stream>>>(agg, (long long)Npad * NF);

    const int nb = Npad / 64;                  // 4 waves x 16 rows per block
    gemm128_kernel<<<nb, 128, 0, stream>>>(h, l1f, x, N);

    const int eb = ((E + 15) / 16 + 7) / 8;    // 8 waves x 16 edges per block
    edge_kernel<<<eb, 256, 0, stream>>>(pos, eidx, w1f, mlp_b1, w2f, mlp_b2,
                                        x, agg, E, NG);

    out_kernel<<<nb, 128, 0, stream>>>(agg, l2f, lin2_b, lf, lin_b, out, N);
}
